// LSTM_71459665871270
// MI455X (gfx1250) — matrix operand order repaired and checked by reading, not compiled, not run on hardware
//
#include <hip/hip_runtime.h>

typedef __bf16 bf16;
typedef __attribute__((ext_vector_type(16))) __bf16 v16bf;
typedef __attribute__((ext_vector_type(8)))  float  v8f;

#define SDIM 256
#define BDIM 16
#define EDIM 512
#define HDIM 1024
#define TDIM 64
#define GDIM 4096      // 4*H gate rows
#define NSTEP 4096     // S*B sequential cell steps per direction

// ---------------------------------------------------------------- helpers
__device__ __forceinline__ v8f wmma_bf16(v16bf a, v16bf b, v8f c) {
  return __builtin_amdgcn_wmma_f32_16x16x32_bf16(false, a, false, b, (short)0, c,
                                                 false, false);
}
__device__ __forceinline__ float sigf(float x) { return 1.0f / (1.0f + __expf(-x)); }

__device__ __forceinline__ void wait_ge(int* p, int v) {
  while (__hip_atomic_load(p, __ATOMIC_ACQUIRE, __HIP_MEMORY_SCOPE_AGENT) < v)
    __builtin_amdgcn_s_sleep(1);
}

// ---------------------------------------------------------------- packing
// A-operand layout (16x32 bf16 tile, wave32):
//   lane 0-15 : row M=lane,     halfs j=0..15 -> K = j
//   lane16-31 : row M=lane-16,  same + 16
__global__ void pack_a(const float* __restrict__ src, bf16* __restrict__ dst,
                       int kTiles, int srcLd, int colOff, int total) {
  int e = blockIdx.x * 256 + threadIdx.x;
  if (e >= total) return;
  int tile = e >> 9, wi = e & 511, lane = wi >> 4, j = wi & 15;
  int rt = tile / kTiles, kt = tile - rt * kTiles;
  int m = rt * 16 + (lane & 15);
  int k = kt * 32 + (lane & 16) + j;
  dst[e] = (bf16)src[(size_t)m * srcLd + colOff + k];
}

// B-operand layout (32x16 bf16 tile): lane holds column n=lane&15,
// halfs j=0..15 -> K = (lane&16) + j   (contiguous in K)
__global__ void pack_b(const float* __restrict__ src, bf16* __restrict__ dst,
                       int kTiles, int srcLd, int colOff, int total) {
  int e = blockIdx.x * 256 + threadIdx.x;
  if (e >= total) return;
  int tile = e >> 9, wi = e & 511, lane = wi >> 4, j = wi & 15;
  int nt = tile / kTiles, kt = tile - nt * kTiles;
  int n = nt * 16 + (lane & 15);
  int k = kt * 32 + (lane & 16) + j;
  dst[e] = (bf16)src[(size_t)n * srcLd + colOff + k];
}

// Pack input x (B,S,E) into A-tiles: m-tile index = physical t, rows = b.
__global__ void pack_x(const float* __restrict__ x, bf16* __restrict__ dst) {
  int e = blockIdx.x * 256 + threadIdx.x;   // exact grid: 256*16*512 elems
  int tile = e >> 9, wi = e & 511, lane = wi >> 4, j = wi & 15;
  int t = tile >> 4, kt = tile & 15;
  int b = lane & 15;
  int k = kt * 32 + (lane & 16) + j;
  dst[e] = (bf16)x[((size_t)b * SDIM + t) * EDIM + k];
}

__global__ void prep_bias(const float* bih0, const float* bhh0,
                          const float* bih1, const float* bhh1,
                          const float* btag, float* b0, float* b1, float* bt) {
  int i = blockIdx.x * 256 + threadIdx.x;
  if (i < GDIM) { b0[i] = bih0[i] + bhh0[i]; b1[i] = bih1[i] + bhh1[i]; }
  if (i < TDIM) bt[i] = btag[i];
}

__global__ void init_flags6(int* a, int* b, int* c, int* d, int* e, int* f) {
  int i = blockIdx.x * 256 + threadIdx.x;
  if (i < NSTEP) { a[i] = 0; b[i] = 0; c[i] = 0; d[i] = 0; e[i] = 0; f[i] = 0; }
}

// ---------------------------------------------------------------- Xproj GEMM
// Xproj[d][s][g] = x_{t(s),b(s)} @ Wih0[:, :E].T + bih0 + bhh0   (bf16 out)
struct GemmParams {
  const bf16* xpack;
  const bf16* wb[2];
  const float* b0[2];
  bf16* xproj[2];
};

__global__ __launch_bounds__(256) void xproj_gemm(GemmParams p) {
  const int lane = threadIdx.x & 31, wave = threadIdx.x >> 5;
  const int bm = blockIdx.x, bn = blockIdx.y, d = blockIdx.z;
  const int nt = bn * 8 + wave;                 // 256 n-tiles total
  const bf16* wbp = p.wb[d] + (size_t)nt * 16 * 512 + lane * 16;
  const int t0 = bm * 4;
  v8f acc[4] = {v8f{}, v8f{}, v8f{}, v8f{}};
  for (int kt = 0; kt < 16; ++kt) {
    v16bf bv = *(const v16bf*)(wbp + (size_t)kt * 512);
#pragma unroll
    for (int a = 0; a < 4; ++a) {
      const int tphys = d ? (SDIM - 1 - (t0 + a)) : (t0 + a);
      v16bf av = *(const v16bf*)(p.xpack + ((size_t)tphys * 16 + kt) * 512 + lane * 16);
      acc[a] = wmma_bf16(av, bv, acc[a]);
    }
  }
  const float bias = p.b0[d][nt * 16 + (lane & 15)];
  const int hi = (lane >> 4) << 3;
  const int ncol = nt * 16 + (lane & 15);
#pragma unroll
  for (int a = 0; a < 4; ++a) {
    const int tau = t0 + a;
#pragma unroll
    for (int r = 0; r < 8; ++r)
      p.xproj[d][(size_t)(tau * 16 + hi + r) * GDIM + ncol] = (bf16)(acc[a][r] + bias);
  }
}

// ---------------------------------------------------------------- recurrence
struct RecParams {
  const bf16* whh0[2]; const bf16* wih0t[2];
  const bf16* wih1[2]; const bf16* whh1[2];
  const bf16* wtag[2]; const bf16* xproj[2];
  const float* b1[2];  const float* btag[2];
  bf16* h0[2]; bf16* h1[2]; bf16* tagb[2]; float* tagf[2];
  int* f0[2]; int* f1[2]; int* ft[2];
};

__global__ __launch_bounds__(256) void lstm_recur(RecParams p) {
  const int bid = blockIdx.x, tid = threadIdx.x;
  const int lane = tid & 31, wave = tid >> 5;
  const int boff = lane & 16;
  __shared__ __align__(32) bf16 sh_v0[HDIM];
  __shared__ __align__(32) bf16 sh_v1[HDIM];
  __shared__ __align__(32) bf16 sh_tag[TDIM];
  __shared__ float sh_g[256];
  float c = 0.0f;

  if (bid < 32) {  // ======================= layer 0 (16 WGs per dir)
    const int d = bid >> 4, w = bid & 15;
    const int mt0 = wave * 2, mt1 = mt0 + 1;
    const int rt0 = (mt0 >> 2) * 64 + w * 4 + (mt0 & 3);
    const int rt1 = (mt1 >> 2) * 64 + w * 4 + (mt1 & 3);
    const bf16* a0 = p.whh0[d]  + (size_t)rt0 * 32 * 512 + lane * 16;
    const bf16* a1 = p.whh0[d]  + (size_t)rt1 * 32 * 512 + lane * 16;
    const bf16* t0 = p.wih0t[d] + (size_t)rt0 * 2 * 512 + lane * 16;
    const bf16* t1 = p.wih0t[d] + (size_t)rt1 * 2 * 512 + lane * 16;
    for (int s = 0; s < NSTEP; ++s) {
      if (tid == 0) {
        if (s >= 16) wait_ge(p.ft[d] + (s - 16), 1);   // prev tag out, same b
        if (s > 0)   wait_ge(p.f0[d] + (s - 1), 16);   // full h0(s-1)
      }
      __syncthreads();
      ((unsigned long long*)sh_v0)[tid] =
          (s > 0) ? ((const unsigned long long*)(p.h0[d] + (size_t)(s - 1) * HDIM))[tid] : 0ull;
      if (tid < 16)
        ((unsigned long long*)sh_tag)[tid] =
            (s >= 16) ? ((const unsigned long long*)(p.tagb[d] + (size_t)(s - 16) * TDIM))[tid] : 0ull;
      __syncthreads();
      v8f acc0 = {}, acc1 = {};
#pragma unroll 4
      for (int kt = 0; kt < 32; ++kt) {               // h(s-1) @ Whh0.T
        v16bf bv = *(const v16bf*)(&sh_v0[kt * 32 + boff]);
        acc0 = wmma_bf16(*(const v16bf*)(a0 + (size_t)kt * 512), bv, acc0);
        acc1 = wmma_bf16(*(const v16bf*)(a1 + (size_t)kt * 512), bv, acc1);
      }
#pragma unroll
      for (int kt = 0; kt < 2; ++kt) {                // tag(t-1,b) @ Wih0[:,E:].T
        v16bf bv = *(const v16bf*)(&sh_tag[kt * 32 + boff]);
        acc0 = wmma_bf16(*(const v16bf*)(t0 + (size_t)kt * 512), bv, acc0);
        acc1 = wmma_bf16(*(const v16bf*)(t1 + (size_t)kt * 512), bv, acc1);
      }
      if ((lane & 15) == 0) {
        const int hi = (lane >> 4) << 3;
#pragma unroll
        for (int r = 0; r < 8; ++r) {
          sh_g[mt0 * 16 + hi + r] = acc0[r];
          sh_g[mt1 * 16 + hi + r] = acc1[r];
        }
      }
      __syncthreads();
      if (tid < 64) {
        const int hu = w * 64 + tid;
        const bf16* xp = p.xproj[d] + (size_t)s * GDIM;   // includes both biases
        float gi = sh_g[tid]       + (float)xp[hu];
        float gf = sh_g[64 + tid]  + (float)xp[HDIM + hu];
        float gg = sh_g[128 + tid] + (float)xp[2 * HDIM + hu];
        float go = sh_g[192 + tid] + (float)xp[3 * HDIM + hu];
        c = sigf(gf) * c + sigf(gi) * tanhf(gg);
        p.h0[d][(size_t)s * HDIM + hu] = (bf16)(sigf(go) * tanhf(c));
      }
      __threadfence();
      __syncthreads();
      if (tid == 0)
        __hip_atomic_fetch_add(p.f0[d] + s, 1, __ATOMIC_RELEASE, __HIP_MEMORY_SCOPE_AGENT);
    }
  } else if (bid < 64) {  // ================ layer 1 (16 WGs per dir)
    const int d = (bid - 32) >> 4, w = (bid - 32) & 15;
    const int mt0 = wave * 2, mt1 = mt0 + 1;
    const int rt0 = (mt0 >> 2) * 64 + w * 4 + (mt0 & 3);
    const int rt1 = (mt1 >> 2) * 64 + w * 4 + (mt1 & 3);
    const bf16* ax0 = p.wih1[d] + (size_t)rt0 * 32 * 512 + lane * 16;
    const bf16* ax1 = p.wih1[d] + (size_t)rt1 * 32 * 512 + lane * 16;
    const bf16* ah0 = p.whh1[d] + (size_t)rt0 * 32 * 512 + lane * 16;
    const bf16* ah1 = p.whh1[d] + (size_t)rt1 * 32 * 512 + lane * 16;
    for (int s = 0; s < NSTEP; ++s) {
      if (tid == 0) {
        wait_ge(p.f0[d] + s, 16);                      // layer-0 output h0(s)
        if (s > 0) wait_ge(p.f1[d] + (s - 1), 16);     // full h1(s-1)
      }
      __syncthreads();
      ((unsigned long long*)sh_v1)[tid] =
          ((const unsigned long long*)(p.h0[d] + (size_t)s * HDIM))[tid];
      ((unsigned long long*)sh_v0)[tid] =
          (s > 0) ? ((const unsigned long long*)(p.h1[d] + (size_t)(s - 1) * HDIM))[tid] : 0ull;
      __syncthreads();
      v8f acc0 = {}, acc1 = {};
#pragma unroll 4
      for (int kt = 0; kt < 32; ++kt) {               // h0(s) @ Wih1.T
        v16bf bv = *(const v16bf*)(&sh_v1[kt * 32 + boff]);
        acc0 = wmma_bf16(*(const v16bf*)(ax0 + (size_t)kt * 512), bv, acc0);
        acc1 = wmma_bf16(*(const v16bf*)(ax1 + (size_t)kt * 512), bv, acc1);
      }
#pragma unroll 4
      for (int kt = 0; kt < 32; ++kt) {               // h1(s-1) @ Whh1.T
        v16bf bv = *(const v16bf*)(&sh_v0[kt * 32 + boff]);
        acc0 = wmma_bf16(*(const v16bf*)(ah0 + (size_t)kt * 512), bv, acc0);
        acc1 = wmma_bf16(*(const v16bf*)(ah1 + (size_t)kt * 512), bv, acc1);
      }
      if ((lane & 15) == 0) {
        const int hi = (lane >> 4) << 3;
#pragma unroll
        for (int r = 0; r < 8; ++r) {
          sh_g[mt0 * 16 + hi + r] = acc0[r];
          sh_g[mt1 * 16 + hi + r] = acc1[r];
        }
      }
      __syncthreads();
      if (tid < 64) {
        const int hu = w * 64 + tid;
        const float* bb = p.b1[d];
        float gi = sh_g[tid]       + bb[hu];
        float gf = sh_g[64 + tid]  + bb[HDIM + hu];
        float gg = sh_g[128 + tid] + bb[2 * HDIM + hu];
        float go = sh_g[192 + tid] + bb[3 * HDIM + hu];
        c = sigf(gf) * c + sigf(gi) * tanhf(gg);
        p.h1[d][(size_t)s * HDIM + hu] = (bf16)(sigf(go) * tanhf(c));
      }
      __threadfence();
      __syncthreads();
      if (tid == 0)
        __hip_atomic_fetch_add(p.f1[d] + s, 1, __ATOMIC_RELEASE, __HIP_MEMORY_SCOPE_AGENT);
    }
  } else {  // ============================== tag head (1 WG per dir)
    const int d = bid - 64;
    for (int s = 0; s < NSTEP; ++s) {
      if (tid == 0) wait_ge(p.f1[d] + s, 16);
      __syncthreads();
      ((unsigned long long*)sh_v0)[tid] =
          ((const unsigned long long*)(p.h1[d] + (size_t)s * HDIM))[tid];
      __syncthreads();
      if (wave < 4) {                                 // Wtag: 64x1024 -> 4 m-tiles
        const bf16* ap = p.wtag[d] + (size_t)wave * 32 * 512 + lane * 16;
        v8f acc = {};
#pragma unroll 4
        for (int kt = 0; kt < 32; ++kt) {
          v16bf bv = *(const v16bf*)(&sh_v0[kt * 32 + boff]);
          acc = wmma_bf16(*(const v16bf*)(ap + (size_t)kt * 512), bv, acc);
        }
        if ((lane & 15) == 0) {
          const int hi = (lane >> 4) << 3;
#pragma unroll
          for (int r = 0; r < 8; ++r) sh_g[wave * 16 + hi + r] = acc[r];
        }
      }
      __syncthreads();
      if (tid < 64) {
        float v = sh_g[tid] + p.btag[d][tid];
        p.tagf[d][(size_t)s * TDIM + tid] = v;
        p.tagb[d][(size_t)s * TDIM + tid] = (bf16)v;
      }
      __threadfence();
      __syncthreads();
      if (tid == 0)
        __hip_atomic_fetch_add(p.ft[d] + s, 1, __ATOMIC_RELEASE, __HIP_MEMORY_SCOPE_AGENT);
    }
  }
}

// ---------------------------------------------------------------- combine
__global__ void combine_out(const float* __restrict__ tf, const float* __restrict__ tb,
                            float* __restrict__ out) {
  const int blk = blockIdx.x;
  const int t = blk & 255, b = blk >> 8;
  const int tid = threadIdx.x;  // 64
  const int s0 = t * 16 + b, s1 = (SDIM - 1 - t) * 16 + b;
  float v = tf[(size_t)s0 * TDIM + tid] + tb[(size_t)s1 * TDIM + tid];
  __shared__ float red[TDIM];
  red[tid] = v; __syncthreads();
  for (int o = 32; o > 0; o >>= 1) {
    if (tid < o) red[tid] = fmaxf(red[tid], red[tid + o]);
    __syncthreads();
  }
  float mx = red[0]; __syncthreads();
  red[tid] = __expf(v - mx); __syncthreads();
  for (int o = 32; o > 0; o >>= 1) {
    if (tid < o) red[tid] += red[tid + o];
    __syncthreads();
  }
  out[((size_t)b * SDIM + t) * TDIM + tid] = v - mx - __logf(red[0]);
}

// ---------------------------------------------------------------- launch
extern "C" void kernel_launch(void* const* d_in, const int* in_sizes, int n_in,
                              void* d_out, int out_size, void* d_ws, size_t ws_size,
                              hipStream_t stream) {
  (void)in_sizes; (void)n_in; (void)out_size; (void)ws_size;
  const float* x = (const float*)d_in[0];
  const float* Wih0[2] = {(const float*)d_in[1],  (const float*)d_in[11]};
  const float* Whh0[2] = {(const float*)d_in[2],  (const float*)d_in[12]};
  const float* bih0[2] = {(const float*)d_in[3],  (const float*)d_in[13]};
  const float* bhh0[2] = {(const float*)d_in[4],  (const float*)d_in[14]};
  const float* Wih1[2] = {(const float*)d_in[5],  (const float*)d_in[15]};
  const float* Whh1[2] = {(const float*)d_in[6],  (const float*)d_in[16]};
  const float* bih1[2] = {(const float*)d_in[7],  (const float*)d_in[17]};
  const float* bhh1[2] = {(const float*)d_in[8],  (const float*)d_in[18]};
  const float* Wtag[2] = {(const float*)d_in[9],  (const float*)d_in[19]};
  const float* btag[2] = {(const float*)d_in[10], (const float*)d_in[20]};

  char* ws = (char*)d_ws;
  size_t off = 0;
  auto alloc = [&](size_t bytes) {
    size_t o = off; off = (off + bytes + 255) & ~(size_t)255; return o;
  };

  bf16 *wxp[2], *whh0p[2], *wt0p[2], *wih1p[2], *whh1p[2], *wtagp[2];
  bf16 *xprojp[2], *h0p[2], *h1p[2], *tagbp[2];
  float *b0p[2], *b1p[2], *btp[2], *tagfp[2];
  int *f0p[2], *f1p[2], *ftp[2];
  for (int d = 0; d < 2; ++d) {
    wxp[d]    = (bf16*)(ws + alloc((size_t)GDIM * EDIM * 2));
    whh0p[d]  = (bf16*)(ws + alloc((size_t)GDIM * HDIM * 2));
    wt0p[d]   = (bf16*)(ws + alloc((size_t)GDIM * TDIM * 2));
    wih1p[d]  = (bf16*)(ws + alloc((size_t)GDIM * HDIM * 2));
    whh1p[d]  = (bf16*)(ws + alloc((size_t)GDIM * HDIM * 2));
    wtagp[d]  = (bf16*)(ws + alloc((size_t)TDIM * HDIM * 2));
    b0p[d]    = (float*)(ws + alloc(GDIM * 4));
    b1p[d]    = (float*)(ws + alloc(GDIM * 4));
    btp[d]    = (float*)(ws + alloc(TDIM * 4));
    xprojp[d] = (bf16*)(ws + alloc((size_t)NSTEP * GDIM * 2));
    h0p[d]    = (bf16*)(ws + alloc((size_t)NSTEP * HDIM * 2));
    h1p[d]    = (bf16*)(ws + alloc((size_t)NSTEP * HDIM * 2));
    tagbp[d]  = (bf16*)(ws + alloc((size_t)NSTEP * TDIM * 2));
    tagfp[d]  = (float*)(ws + alloc((size_t)NSTEP * TDIM * 4));
    f0p[d]    = (int*)(ws + alloc(NSTEP * 4));
    f1p[d]    = (int*)(ws + alloc(NSTEP * 4));
    ftp[d]    = (int*)(ws + alloc(NSTEP * 4));
  }
  bf16* xpackp = (bf16*)(ws + alloc((size_t)SDIM * 16 * 512 * 2));

  init_flags6<<<16, 256, 0, stream>>>(f0p[0], f0p[1], f1p[0], f1p[1], ftp[0], ftp[1]);

  for (int d = 0; d < 2; ++d) {
    pack_b<<<8192, 256, 0, stream>>>(Wih0[d], wxp[d], 16, EDIM + TDIM, 0, 256 * 16 * 512);
    pack_a<<<16384, 256, 0, stream>>>(Whh0[d], whh0p[d], 32, HDIM, 0, 256 * 32 * 512);
    pack_a<<<1024, 256, 0, stream>>>(Wih0[d], wt0p[d], 2, EDIM + TDIM, EDIM, 256 * 2 * 512);
    pack_a<<<16384, 256, 0, stream>>>(Wih1[d], wih1p[d], 32, HDIM, 0, 256 * 32 * 512);
    pack_a<<<16384, 256, 0, stream>>>(Whh1[d], whh1p[d], 32, HDIM, 0, 256 * 32 * 512);
    pack_a<<<256, 256, 0, stream>>>(Wtag[d], wtagp[d], 32, HDIM, 0, 4 * 32 * 512);
    prep_bias<<<16, 256, 0, stream>>>(bih0[d], bhh0[d], bih1[d], bhh1[d], btag[d],
                                      b0p[d], b1p[d], btp[d]);
  }
  pack_x<<<8192, 256, 0, stream>>>(x, xpackp);

  GemmParams gp;
  gp.xpack = xpackp;
  for (int d = 0; d < 2; ++d) { gp.wb[d] = wxp[d]; gp.b0[d] = b0p[d]; gp.xproj[d] = xprojp[d]; }
  xproj_gemm<<<dim3(64, 32, 2), 256, 0, stream>>>(gp);

  RecParams rp;
  for (int d = 0; d < 2; ++d) {
    rp.whh0[d] = whh0p[d]; rp.wih0t[d] = wt0p[d];
    rp.wih1[d] = wih1p[d]; rp.whh1[d]  = whh1p[d];
    rp.wtag[d] = wtagp[d]; rp.xproj[d] = xprojp[d];
    rp.b1[d] = b1p[d]; rp.btag[d] = btp[d];
    rp.h0[d] = h0p[d]; rp.h1[d] = h1p[d];
    rp.tagb[d] = tagbp[d]; rp.tagf[d] = tagfp[d];
    rp.f0[d] = f0p[d]; rp.f1[d] = f1p[d]; rp.ft[d] = ftp[d];
  }
  lstm_recur<<<66, 256, 0, stream>>>(rp);

  combine_out<<<4096, 64, 0, stream>>>(tagfp[0], tagfp[1], (float*)d_out);
}
